// Cloud_Mamba_Reg_2199023255679
// MI455X (gfx1250) — matrix operand order repaired and checked by reading, not compiled, not run on hardware
//
#include <hip/hip_runtime.h>
#include <hip/hip_bf16.h>
#include <math.h>

// ---------------- problem constants ----------------
#define Bn   8
#define Gn   512
#define Dd   384
#define DIc  768
#define Sn   16
#define Rn   24
#define Kc   4
#define M4   (Bn*Gn)        // 4096 rows
#define LDXZ (2*DIc)        // 1536
#define LDXD (Rn + 2*Sn)    // 56
#define NT   4              // N-tiles per wave in the WMMA GEMM

typedef __attribute__((ext_vector_type(16))) __bf16 v16bf;
typedef __attribute__((ext_vector_type(8)))  float  v8f;

// ---------------- bitonic argsort over G=512 keys ----------------
__global__ void argsort_kernel(const float* __restrict__ center, int comp,
                               int* __restrict__ order, int* __restrict__ inv) {
    __shared__ float key[Gn];
    __shared__ int   idx[Gn];
    const int b = blockIdx.x, t = threadIdx.x;
    key[t] = center[((size_t)b * Gn + t) * 3 + comp];
    idx[t] = t;
    __syncthreads();
    for (int k = 2; k <= Gn; k <<= 1) {
        for (int j = k >> 1; j > 0; j >>= 1) {
            int ixj = t ^ j;
            if (ixj > t) {
                float ka = key[t], kb = key[ixj];
                int   ia = idx[t], ib = idx[ixj];
                bool asc = ((t & k) == 0);
                bool gt  = (ka > kb) || (ka == kb && ia > ib);
                if (gt == asc) { key[t] = kb; key[ixj] = ka; idx[t] = ib; idx[ixj] = ia; }
            }
            __syncthreads();
        }
    }
    order[b * Gn + t] = idx[t];
    inv[b * Gn + idx[t]] = t;
}

// ---------------- row gather (also scatter-by-inverse into cat) -------------
__global__ void gather_rows_kernel(const float* __restrict__ in, const int* __restrict__ idx,
                                   float* __restrict__ out, int out_ld, int out_off, int total) {
    int tid = blockIdx.x * blockDim.x + threadIdx.x;
    if (tid >= total) return;
    int bg = tid / Dd, d = tid % Dd;
    int b = bg / Gn;
    int src = b * Gn + idx[bg];
    out[(size_t)bg * out_ld + out_off + d] = in[(size_t)src * Dd + d];
}

// ---------------- LayerNorm: one wave per row -------------------------------
__global__ void ln_kernel(const float* __restrict__ x, const float* __restrict__ w,
                          const float* __restrict__ bb, float* __restrict__ out, int rows) {
    int lane = threadIdx.x & 31;
    int row  = blockIdx.x * (blockDim.x >> 5) + (threadIdx.x >> 5);
    if (row >= rows) return;
    const float* xr = x + (size_t)row * Dd;
    float s = 0.f, sq = 0.f;
    for (int d = lane; d < Dd; d += 32) { float v = xr[d]; s += v; sq += v * v; }
    for (int o = 16; o; o >>= 1) { s += __shfl_xor(s, o, 32); sq += __shfl_xor(sq, o, 32); }
    float mean = s / (float)Dd;
    float var  = sq / (float)Dd - mean * mean;
    float rstd = rsqrtf(var + 1e-5f);
    float* orow = out + (size_t)row * Dd;
    for (int d = lane; d < Dd; d += 32) orow[d] = (xr[d] - mean) * rstd * w[d] + bb[d];
}

// ---- pack 8 f32 (two float4) into half of a v16bf fragment -----------------
__device__ __forceinline__ void pack8(v16bf& f, int base, float4 lo, float4 hi) {
    f[base + 0] = (__bf16)lo.x; f[base + 1] = (__bf16)lo.y;
    f[base + 2] = (__bf16)lo.z; f[base + 3] = (__bf16)lo.w;
    f[base + 4] = (__bf16)hi.x; f[base + 5] = (__bf16)hi.y;
    f[base + 6] = (__bf16)hi.z; f[base + 7] = (__bf16)hi.w;
}

// ---------------- WMMA bf16 GEMM: D = act(A * Bt^T + bias + res) ------------
// A: [M x lda] f32 row-major. Bt: [N x Kd] f32 row-major. Each wave: 16 rows x
// NT*16 cols, A fragment reused across NT B fragments. Lanes with n>=N clamp
// their B-row pointer to row 0 (garbage flows only into never-stored columns,
// so EXEC stays all-ones around v_wmma). K runs are 8-float aligned (all Kd
// here are multiples of 8), so the K-tail guard is per-run, fast path has none.
__global__ void wmma_gemm_kernel(const float* __restrict__ A, const float* __restrict__ Bt,
                                 const float* __restrict__ bias, const float* __restrict__ res,
                                 float* __restrict__ Dst,
                                 int M, int N, int Kd, int lda, int ldd,
                                 int act, int flip) {
    const int tn  = (N + 15) >> 4;            // 16-wide N tiles
    const int tng = (tn + NT - 1) / NT;       // tile groups per row of tiles
    const int tm  = (M + 15) >> 4;
    int wave = blockIdx.x * (blockDim.x >> 5) + (threadIdx.x >> 5);
    if (wave >= tm * tng) return;
    const int tM = wave / tng, tG = wave % tng;
    const int lane = threadIdx.x & 31;
    const int hgrp = lane >> 4;               // half-wave K-group (ISA layout)
    const int lm   = lane & 15;

    const float* arow = A + (size_t)(tM * 16 + lm) * lda;
    const float* brow[NT];
#pragma unroll
    for (int u = 0; u < NT; ++u) {
        int nB = (tG * NT + u) * 16 + lm;
        brow[u] = Bt + (size_t)(nB < N ? nB : 0) * Kd;
    }

    v8f acc[NT];
#pragma unroll
    for (int u = 0; u < NT; ++u) acc[u] = (v8f){0.f,0.f,0.f,0.f,0.f,0.f,0.f,0.f};

    const int kfull = Kd & ~31;               // K%32==0 fast path bound
    int k0 = 0;
    for (; k0 < kfull; k0 += 32) {
        // A fragment: runs at k0+8*hgrp and k0+16+8*hgrp (8 floats each)
        const float4* pa0 = (const float4*)(arow + k0 + 8 * hgrp);
        const float4* pa1 = (const float4*)(arow + k0 + 16 + 8 * hgrp);
        float4 a0 = pa0[0], a1 = pa0[1], a2 = pa1[0], a3 = pa1[1];
        __builtin_prefetch(arow + k0 + 32, 0, 1);
        v16bf af;
        pack8(af, 0, a0, a1);
        pack8(af, 8, a2, a3);
#pragma unroll
        for (int u = 0; u < NT; ++u) {
            // B fragment: runs at k0+8*hgrp and k0+16+8*hgrp (8 floats each)
            const float4* pb0 = (const float4*)(brow[u] + k0 + 8 * hgrp);
            const float4* pb1 = (const float4*)(brow[u] + k0 + 16 + 8 * hgrp);
            float4 b0 = pb0[0], b1 = pb0[1], b2 = pb1[0], b3 = pb1[1];
            __builtin_prefetch(brow[u] + k0 + 32, 0, 1);
            v16bf bfr;
            pack8(bfr, 0, b0, b1);
            pack8(bfr, 8, b2, b3);
            acc[u] = __builtin_amdgcn_wmma_f32_16x16x32_bf16(false, af, false, bfr,
                                                             (short)0, acc[u], false, false);
        }
    }
    if (k0 < Kd) {                            // K tail (Kd%32!=0, e.g. Kd=24)
        const float4 z4 = {0.f, 0.f, 0.f, 0.f};
        auto ld8 = [&](const float* p, int k, float4& lo, float4& hi) {
            if (k < Kd) { const float4* q = (const float4*)(p + k); lo = q[0]; hi = q[1]; }
            else        { lo = z4; hi = z4; }
        };
        float4 a0, a1, a2, a3;
        ld8(arow, k0 + 8 * hgrp,      a0, a1);
        ld8(arow, k0 + 16 + 8 * hgrp, a2, a3);
        v16bf af;
        pack8(af, 0, a0, a1);
        pack8(af, 8, a2, a3);
#pragma unroll
        for (int u = 0; u < NT; ++u) {
            float4 b0, b1, b2, b3;
            ld8(brow[u], k0 + 8 * hgrp,      b0, b1);
            ld8(brow[u], k0 + 16 + 8 * hgrp, b2, b3);
            v16bf bfr;
            pack8(bfr, 0, b0, b1);
            pack8(bfr, 8, b2, b3);
            acc[u] = __builtin_amdgcn_wmma_f32_16x16x32_bf16(false, af, false, bfr,
                                                             (short)0, acc[u], false, false);
        }
    }

#pragma unroll
    for (int u = 0; u < NT; ++u) {
        int nB = (tG * NT + u) * 16 + lm;
        bool nok = nB < N;
        float bval = (bias && nok) ? bias[nB] : 0.0f;
#pragma unroll
        for (int v = 0; v < 8; ++v) {
            int m = tM * 16 + hgrp * 8 + v;   // C/D layout: M = 8*half + vgpr
            float val = acc[u][v] + bval;
            if (res && nok) val += res[(size_t)m * ldd + nB];
            if (act == 1)      val = 0.5f * val * (1.0f + erff(val * 0.70710678118654752f));
            else if (act == 2) val = (val > 20.0f) ? val : log1pf(__expf(val));
            int orow = m;
            if (flip) orow = (m & ~(Gn - 1)) | ((Gn - 1) - (m & (Gn - 1)));
            if (nok && m < M) Dst[(size_t)orow * ldd + nB] = val;
        }
    }
}

// ---------------- depthwise causal conv (K=4) + SiLU ------------------------
__global__ void conv_silu_kernel(const float* __restrict__ xz, const float* __restrict__ cw,
                                 const float* __restrict__ cb, float* __restrict__ xc, int total) {
    int tid = blockIdx.x * blockDim.x + threadIdx.x;
    if (tid >= total) return;
    int c  = tid % DIc;
    int bg = tid / DIc;
    int g  = bg % Gn, b = bg / Gn;
    float acc = cb[c];
#pragma unroll
    for (int j = 0; j < Kc; ++j) {
        int gs = g + j - (Kc - 1);
        if (gs >= 0) acc += xz[((size_t)(b * Gn + gs)) * LDXZ + c] * cw[c * Kc + j];
    }
    xc[tid] = acc / (1.0f + __expf(-acc));   // silu
}

// ---------------- selective scan: one lane per (b, channel) -----------------
__global__ void scan_kernel(const float* __restrict__ dt, const float* __restrict__ xc,
                            const float* __restrict__ xdbl, const float* __restrict__ A_log,
                            const float* __restrict__ Dp, float* __restrict__ y) {
    int c = blockIdx.y * blockDim.x + threadIdx.x;
    int b = blockIdx.x;
    if (c >= DIc) return;
    float A[Sn];
#pragma unroll
    for (int s = 0; s < Sn; ++s) A[s] = -__expf(A_log[c * Sn + s]);
    const float Dc = Dp[c];
    float h[Sn];
#pragma unroll
    for (int s = 0; s < Sn; ++s) h[s] = 0.f;
    for (int t = 0; t < Gn; ++t) {
        size_t row = (size_t)b * Gn + t;
        float dtc = dt[row * DIc + c];
        float xcv = xc[row * DIc + c];
        float dx  = dtc * xcv;
        const float* xr = xdbl + row * LDXD;
        float accv = 0.f;
#pragma unroll
        for (int s = 0; s < Sn; ++s) {
            float dA = __expf(dtc * A[s]);
            h[s] = dA * h[s] + dx * xr[Rn + s];          // Bs
            accv += h[s] * xr[Rn + Sn + s];              // Cs
        }
        y[row * DIc + c] = accv + xcv * Dc;
    }
}

// ---------------- y *= silu(z)  (z lives in xz[:, DI:2DI]) ------------------
__global__ void gate_kernel(float* __restrict__ y, const float* __restrict__ xz, int total) {
    int tid = blockIdx.x * blockDim.x + threadIdx.x;
    if (tid >= total) return;
    int row = tid / DIc, c = tid % DIc;
    float z = xz[(size_t)row * LDXZ + DIc + c];
    y[tid] *= z / (1.0f + __expf(-z));
}

// ---------------- host orchestration ----------------------------------------
static inline void launch_gemm(const float* A, const float* Bt, const float* bias,
                               const float* res, float* Dst, int M, int N, int Kd,
                               int lda, int ldd, int act, int flip, hipStream_t s) {
    int tn  = (N + 15) / 16;
    int tng = (tn + NT - 1) / NT;
    int waves = ((M + 15) / 16) * tng;
    int blocks = (waves + 7) / 8;   // 8 waves / 256-thread block
    wmma_gemm_kernel<<<blocks, 256, 0, s>>>(A, Bt, bias, res, Dst, M, N, Kd, lda, ldd, act, flip);
}

extern "C" void kernel_launch(void* const* d_in, const int* in_sizes, int n_in,
                              void* d_out, int out_size, void* d_ws, size_t ws_size,
                              hipStream_t stream) {
    (void)in_sizes; (void)n_in; (void)out_size; (void)ws_size;
    const float* center   = (const float*)d_in[0];
    const float* gin      = (const float*)d_in[1];
    const float* norm_w   = (const float*)d_in[2];
    const float* norm_b   = (const float*)d_in[3];
    const float* inproj_w = (const float*)d_in[4];
    const float* conv_w   = (const float*)d_in[5];
    const float* conv_b   = (const float*)d_in[6];
    const float* xproj_w  = (const float*)d_in[7];
    const float* dtproj_w = (const float*)d_in[8];
    const float* dtproj_b = (const float*)d_in[9];
    const float* A_log    = (const float*)d_in[10];
    const float* Dp       = (const float*)d_in[11];
    const float* outproj_w= (const float*)d_in[12];
    const float* normf_w  = (const float*)d_in[13];
    const float* normf_b  = (const float*)d_in[14];
    const float* down_w1  = (const float*)d_in[15];
    const float* down_b1  = (const float*)d_in[16];
    const float* down_w2  = (const float*)d_in[17];
    const float* down_b2  = (const float*)d_in[18];
    float* outp = (float*)d_out;

    // workspace carve-up (256B aligned)
    char* ws = (char*)d_ws;
    size_t off = 0;
    auto carve = [&](size_t bytes) { char* p = ws + off; off = (off + bytes + 255) & ~(size_t)255; return p; };
    int*   ord  = (int*)  carve(3 * Bn * Gn * sizeof(int));
    int*   inv  = (int*)  carve(3 * Bn * Gn * sizeof(int));
    float* hA   = (float*)carve((size_t)M4 * Dd * 4);
    float* hB   = (float*)carve((size_t)M4 * Dd * 4);
    float* lnout= (float*)carve((size_t)M4 * Dd * 4);
    float* xz   = (float*)carve((size_t)M4 * LDXZ * 4);
    float* xc   = (float*)carve((size_t)M4 * DIc * 4);
    float* xdbl = (float*)carve((size_t)M4 * LDXD * 4);
    float* dtb  = (float*)carve((size_t)M4 * DIc * 4);
    float* ybuf = (float*)carve((size_t)M4 * DIc * 4);
    float* cat  = (float*)carve((size_t)M4 * 3 * Dd * 4);
    float* h1   = (float*)carve((size_t)M4 * Dd * 4);

    // 1) argsorts: mixer m uses coordinate component 2-m (z,y,x)
    for (int m = 0; m < 3; ++m)
        argsort_kernel<<<Bn, Gn, 0, stream>>>(center, 2 - m, ord + m * Bn * Gn, inv + m * Bn * Gn);

    const int totD  = M4 * Dd;
    const int totDI = M4 * DIc;
    const int ln_blocks = (M4 + 7) / 8;

    for (int m = 0; m < 3; ++m) {
        gather_rows_kernel<<<(totD + 255) / 256, 256, 0, stream>>>(gin, ord + m * Bn * Gn, hA, Dd, 0, totD);
        float* cur = hA; float* nxt = hB;
        for (int lb = 0; lb < 4; ++lb) {          // NL=2 layers x 2 directions
            int i = (m * 2) * 2 + lb;             // (m*NL + l)*2 + dir, NL=2
            ln_kernel<<<ln_blocks, 256, 0, stream>>>(cur, norm_w + i * Dd, norm_b + i * Dd, lnout, M4);
            // inproj: [4096,1536] = lnout[4096,384] x W[1536,384]^T
            launch_gemm(lnout, inproj_w + (size_t)i * LDXZ * Dd, nullptr, nullptr,
                        xz, M4, LDXZ, Dd, Dd, LDXZ, 0, 0, stream);
            conv_silu_kernel<<<(totDI + 255) / 256, 256, 0, stream>>>(
                xz, conv_w + (size_t)i * DIc * Kc, conv_b + (size_t)i * DIc, xc, totDI);
            // xproj: [4096,56] = xc x W[56,768]^T
            launch_gemm(xc, xproj_w + (size_t)i * LDXD * DIc, nullptr, nullptr,
                        xdbl, M4, LDXD, DIc, DIc, LDXD, 0, 0, stream);
            // dtproj + bias + softplus: [4096,768] = xdbl[:, :24] x W[768,24]^T
            launch_gemm(xdbl, dtproj_w + (size_t)i * DIc * Rn, dtproj_b + (size_t)i * DIc, nullptr,
                        dtb, M4, DIc, Rn, LDXD, DIc, 2, 0, stream);
            scan_kernel<<<dim3(Bn, DIc / 256), 256, 0, stream>>>(
                dtb, xc, xdbl, A_log + (size_t)i * DIc * Sn, Dp + (size_t)i * DIc, ybuf);
            gate_kernel<<<(totDI + 255) / 256, 256, 0, stream>>>(ybuf, xz, totDI);
            // outproj + residual, store flipped along sequence
            launch_gemm(ybuf, outproj_w + (size_t)i * Dd * DIc, nullptr, cur,
                        nxt, M4, Dd, DIc, DIc, Dd, 0, 1, stream);
            float* t_ = cur; cur = nxt; nxt = t_;
        }
        ln_kernel<<<ln_blocks, 256, 0, stream>>>(cur, normf_w + m * Dd, normf_b + m * Dd, lnout, M4);
        gather_rows_kernel<<<(totD + 255) / 256, 256, 0, stream>>>(
            lnout, inv + m * Bn * Gn, cat, 3 * Dd, m * Dd, totD);
    }

    // down-proj MLP: gelu(cat @ w1^T + b1) @ w2^T + b2
    launch_gemm(cat, down_w1, down_b1, nullptr, h1, M4, Dd, 3 * Dd, 3 * Dd, Dd, 1, 0, stream);
    launch_gemm(h1, down_w2, down_b2, nullptr, outp, M4, Dd, Dd, Dd, Dd, 0, 0, stream);
}
